// InfiniteHoloGPT_27281632264900
// MI455X (gfx1250) — matrix operand order repaired and checked by reading, not compiled, not run on hardware
//
#include <hip/hip_runtime.h>
#include <hip/hip_bf16.h>
#include <math.h>

// Problem constants (match the reference).
#define Vv   32000
#define Dd   512
#define Mm   128
#define Ll   4
#define Bb   8
#define Tt   256
#define BT   (Bb*Tt)    // 2048 rows
#define HID  (4*Dd)     // 2048

typedef __attribute__((ext_vector_type(16))) __bf16 v16bf;
typedef __attribute__((ext_vector_type(8)))  __bf16 v8bf;   // one 128-bit load
typedef __attribute__((ext_vector_type(4)))  __bf16 v4bf;
typedef __attribute__((ext_vector_type(8)))  float  v8f;

__device__ __forceinline__ v16bf cat16(v8bf lo, v8bf hi) {
    return __builtin_shufflevector(lo, hi, 0,1,2,3,4,5,6,7,8,9,10,11,12,13,14,15);
}

// ---------------------------------------------------------------------------
// f32 -> bf16 bulk convert (vectorized x4). n4 = n/4.
// ---------------------------------------------------------------------------
__global__ void __launch_bounds__(256)
cvt_bf16_kernel(const float* __restrict__ in, __bf16* __restrict__ out, int n4)
{
    int i = blockIdx.x * 256 + threadIdx.x;
    if (i >= n4) return;
    float4 v = reinterpret_cast<const float4*>(in)[i];
    v4bf o;
    o[0] = (__bf16)v.x; o[1] = (__bf16)v.y; o[2] = (__bf16)v.z; o[3] = (__bf16)v.w;
    reinterpret_cast<v4bf*>(out)[i] = o;
}

// ---------------------------------------------------------------------------
// Embedding gather: x[b,t,:] = center[idx[b,t], :]   (box min/max math cancels)
// ---------------------------------------------------------------------------
__global__ void embed_kernel(const int* __restrict__ idx,
                             const float* __restrict__ center,
                             float* __restrict__ x, int n, int D)
{
    int i = blockIdx.x * blockDim.x + threadIdx.x;
    if (i >= n) return;
    int tkn = i / D;
    int d   = i - tkn * D;
    x[i] = center[(size_t)idx[tkn] * D + d];
}

// ---------------------------------------------------------------------------
// BF16 WMMA NT-GEMM, register-blocked: one wave computes a 32x64 output tile
// (2 row-tiles x 4 col-tiles of 16x16) -> 8 v_wmma per 32-deep K step, fed by
// 12 b128 loads (A fragments reused 4x, B fragments 2x). Operands are bf16 in
// memory; with this code's contiguous k order (same for A and B) each fragment is two contiguous 128-bit loads:
//   A 16x32 : lane(half=l>>4, m=l&15): 16-run at kb+half*16
//   B 32x16 : lane(half, n=l&15):      16-run at kb+half*16
//   C 16x16 f32 : VGPR v -> row v + 8*half, col = lane&15
// Epilogue: +bias -> act(0:none,1:gelu,2:tanh) -> *scale -> +residual,
// written to f32 (Cf) and/or bf16 (Cbf).
// ---------------------------------------------------------------------------
__global__ void __launch_bounds__(256)
gemm_bf16_nt(const __bf16* __restrict__ A, const __bf16* __restrict__ W,
             const float* __restrict__ bias, const float* __restrict__ residual,
             const float* __restrict__ scale_ptr, float scale_mul,
             float* __restrict__ Cf, __bf16* __restrict__ Cbf,
             int rows, int cols, int K, int act)
{
    const int wave = threadIdx.x >> 5;
    const int lane = threadIdx.x & 31;
    const int tilesX = cols >> 6;             // 64-wide wave tiles
    const int tilesY = rows >> 5;             // 32-tall wave tiles
    const int tw = blockIdx.x * 8 + wave;
    if (tw >= tilesX * tilesY) return;        // wave-uniform: EXEC stays all-1s
    const int ty = tw / tilesX;
    const int tx = tw - ty * tilesX;
    const int row0 = ty << 5;
    const int col0 = tx << 6;
    const int half = lane >> 4;
    const int l16  = lane & 15;

    const __bf16* Ar0 = A + (size_t)(row0 + l16) * K;       // rows row0..+15
    const __bf16* Ar1 = Ar0 + (size_t)16 * K;               // rows row0+16..+31
    const __bf16* Wr0 = W + (size_t)(col0 + l16) * K;
    const __bf16* Wr1 = Wr0 + (size_t)16 * K;
    const __bf16* Wr2 = Wr0 + (size_t)32 * K;
    const __bf16* Wr3 = Wr0 + (size_t)48 * K;

    v8f acc[2][4] = {};
    for (int kb = 0; kb < K; kb += 32) {
        const int kaLo = kb + half * 16;      // A fragment: one 16-element run
        const int kaHi = kb + half * 16 + 8;
        const int kbB  = kb + half * 16;      // B fragment: one 16-element run

        v16bf aF[2], bF[4];
        aF[0] = cat16(*reinterpret_cast<const v8bf*>(Ar0 + kaLo),
                      *reinterpret_cast<const v8bf*>(Ar0 + kaHi));
        aF[1] = cat16(*reinterpret_cast<const v8bf*>(Ar1 + kaLo),
                      *reinterpret_cast<const v8bf*>(Ar1 + kaHi));
        bF[0] = cat16(*reinterpret_cast<const v8bf*>(Wr0 + kbB),
                      *reinterpret_cast<const v8bf*>(Wr0 + kbB + 8));
        bF[1] = cat16(*reinterpret_cast<const v8bf*>(Wr1 + kbB),
                      *reinterpret_cast<const v8bf*>(Wr1 + kbB + 8));
        bF[2] = cat16(*reinterpret_cast<const v8bf*>(Wr2 + kbB),
                      *reinterpret_cast<const v8bf*>(Wr2 + kbB + 8));
        bF[3] = cat16(*reinterpret_cast<const v8bf*>(Wr3 + kbB),
                      *reinterpret_cast<const v8bf*>(Wr3 + kbB + 8));

#pragma unroll
        for (int mr = 0; mr < 2; ++mr)
#pragma unroll
            for (int nc = 0; nc < 4; ++nc)
                acc[mr][nc] = __builtin_amdgcn_wmma_f32_16x16x32_bf16(
                    false, aF[mr], false, bF[nc], (short)0, acc[mr][nc], false, false);
    }

    const float scale = scale_mul * (scale_ptr ? scale_ptr[0] : 1.0f);
#pragma unroll
    for (int mr = 0; mr < 2; ++mr) {
#pragma unroll
        for (int nc = 0; nc < 4; ++nc) {
            const int col = col0 + nc * 16 + l16;
            const float bval = bias ? bias[col] : 0.0f;
#pragma unroll
            for (int v = 0; v < 8; ++v) {
                const int row = row0 + mr * 16 + v + 8 * half;
                float val = acc[mr][nc][v] + bval;
                if (act == 1)      val = 0.5f * val * (1.0f + erff(val * 0.70710678118654752f));
                else if (act == 2) val = tanhf(val);
                val *= scale;
                const size_t o = (size_t)row * cols + col;
                if (residual) val += residual[o];
                if (Cf)  Cf[o]  = val;
                if (Cbf) Cbf[o] = (__bf16)val;
            }
        }
    }
}

// ---------------------------------------------------------------------------
// LayerNorm: one block per row of width D. Writes f32 and optional bf16 copy.
// ---------------------------------------------------------------------------
__global__ void __launch_bounds__(256)
layernorm_kernel(const float* __restrict__ x, const float* __restrict__ g,
                 const float* __restrict__ b, float* __restrict__ out,
                 __bf16* __restrict__ out_bf, int D)
{
    __shared__ float s_sum[256], s_sq[256];
    const int row = blockIdx.x;
    const float* xr = x + (size_t)row * D;
    float sum = 0.f, sq = 0.f;
    for (int i = threadIdx.x; i < D; i += 256) { float v = xr[i]; sum += v; sq += v * v; }
    s_sum[threadIdx.x] = sum; s_sq[threadIdx.x] = sq;
    __syncthreads();
    for (int s = 128; s > 0; s >>= 1) {
        if (threadIdx.x < s) {
            s_sum[threadIdx.x] += s_sum[threadIdx.x + s];
            s_sq[threadIdx.x]  += s_sq[threadIdx.x + s];
        }
        __syncthreads();
    }
    const float mean = s_sum[0] / (float)D;
    const float var  = s_sq[0] / (float)D - mean * mean;
    const float rstd = rsqrtf(var + 1e-5f);
    for (int i = threadIdx.x; i < D; i += 256) {
        const float o = (xr[i] - mean) * rstd * g[i] + b[i];
        if (out)    out[(size_t)row * D + i]    = o;
        if (out_bf) out_bf[(size_t)row * D + i] = (__bf16)o;
    }
}

// ---------------------------------------------------------------------------
// Row-wise L2 normalize (in place), M == 128, one block of 128 per row.
// ---------------------------------------------------------------------------
__global__ void __launch_bounds__(128)
l2norm_kernel(float* __restrict__ p)
{
    __shared__ float s[128];
    const int row = blockIdx.x;
    const float v = p[(size_t)row * Mm + threadIdx.x];
    s[threadIdx.x] = v * v;
    __syncthreads();
    for (int st = 64; st > 0; st >>= 1) {
        if (threadIdx.x < st) s[threadIdx.x] += s[threadIdx.x + st];
        __syncthreads();
    }
    const float n = sqrtf(s[0]);
    p[(size_t)row * Mm + threadIdx.x] = v / fmaxf(n, 1e-12f);
}

// ---------------------------------------------------------------------------
// Gates: gw = sigmoid(xn.wgw + bgw) * 1.0 ; gf = sigmoid(xn.wgf + bgf) * 0.9
// (rarity == 1 in the reference). One block of 128 per row.
// ---------------------------------------------------------------------------
__global__ void __launch_bounds__(128)
gate_kernel(const float* __restrict__ xn, const float* __restrict__ wgw,
            const float* __restrict__ bgw, const float* __restrict__ wgf,
            const float* __restrict__ bgf, float* __restrict__ gw,
            float* __restrict__ gf, int D)
{
    __shared__ float sw[128], sf[128];
    const int row = blockIdx.x;
    float dw = 0.f, df = 0.f;
    for (int i = threadIdx.x; i < D; i += 128) {
        const float v = xn[(size_t)row * D + i];
        dw += v * wgw[i];
        df += v * wgf[i];
    }
    sw[threadIdx.x] = dw; sf[threadIdx.x] = df;
    __syncthreads();
    for (int s = 64; s > 0; s >>= 1) {
        if (threadIdx.x < s) { sw[threadIdx.x] += sw[threadIdx.x + s];
                               sf[threadIdx.x] += sf[threadIdx.x + s]; }
        __syncthreads();
    }
    if (threadIdx.x == 0) {
        gw[row] = 1.0f / (1.0f + expf(-(sw[0] + bgw[0])));
        gf[row] = 0.9f / (1.0f + expf(-(sf[0] + bgf[0])));
    }
}

// ---------------------------------------------------------------------------
// Sequential associative-memory scan. One block per batch element.
// mem[128x128] f32 held in 64KB LDS (WGP has 320KB). Per step:
//   readout = mem @ q_t  (before update), mem = decay*mem + beta * v_t k_t^T
// Readouts written as bf16 (feed the wo WMMA GEMM); final mem -> d_out states.
// ---------------------------------------------------------------------------
__global__ void __launch_bounds__(256)
scan_kernel(const float* __restrict__ k, const float* __restrict__ q,
            const float* __restrict__ v, const float* __restrict__ gw,
            const float* __restrict__ gf, __bf16* __restrict__ reads_bf,
            float* __restrict__ states, int T)
{
    __shared__ float mem[Mm * Mm];
    __shared__ float qs[Mm], ks[Mm], vs[Mm];
    __shared__ float red[256];
    __shared__ float beta_s, decay_s;

    const int b = blockIdx.x, tid = threadIdx.x;
    for (int e = tid; e < Mm * Mm; e += 256) mem[e] = 0.0f;

    const int i  = tid >> 1;          // memory row owned for matvec/update
    const int jb = (tid & 1) * 64;    // half of the row

    for (int t = 0; t < T; ++t) {
        const size_t r = (size_t)(b * T + t) * Mm;
        if (tid < Mm) { qs[tid] = q[r + tid]; ks[tid] = k[r + tid]; vs[tid] = v[r + tid]; }
        if (tid == 0) { beta_s = gw[b * T + t]; decay_s = gf[b * T + t]; }
        __syncthreads();

        float p = 0.f;
#pragma unroll 8
        for (int j = 0; j < 64; ++j) p += mem[i * Mm + jb + j] * qs[jb + j];
        red[tid] = p;
        const float beta = beta_s, decay = decay_s, vi = vs[i];
        __syncthreads();

        if ((tid & 1) == 0) reads_bf[r + i] = (__bf16)(red[tid] + red[tid + 1]);

#pragma unroll 8
        for (int j = 0; j < 64; ++j) {
            const int e = i * Mm + jb + j;
            mem[e] = decay * mem[e] + beta * vi * ks[jb + j];
        }
        __syncthreads();
    }
    for (int e = tid; e < Mm * Mm; e += 256)
        states[(size_t)b * Mm * Mm + e] = mem[e];
}

// ---------------------------------------------------------------------------
extern "C" void kernel_launch(void* const* d_in, const int* in_sizes, int n_in,
                              void* d_out, int out_size, void* d_ws, size_t ws_size,
                              hipStream_t stream)
{
    (void)in_sizes; (void)n_in; (void)out_size; (void)ws_size;

    const int*   idx    = (const int*)  d_in[0];
    const float* center = (const float*)d_in[1];
    /* d_in[2] = offset: unused — box midpoint cancels softplus */
    const float* wk   = (const float*)d_in[3];
    const float* bk   = (const float*)d_in[4];
    const float* wq   = (const float*)d_in[5];
    const float* bq   = (const float*)d_in[6];
    const float* wv   = (const float*)d_in[7];
    const float* bv   = (const float*)d_in[8];
    const float* wo   = (const float*)d_in[9];
    const float* bo   = (const float*)d_in[10];
    const float* wgw  = (const float*)d_in[11];
    const float* bgw  = (const float*)d_in[12];
    const float* wgf  = (const float*)d_in[13];
    const float* bgf  = (const float*)d_in[14];
    const float* ln1g = (const float*)d_in[15];
    const float* ln1b = (const float*)d_in[16];
    const float* ln2g = (const float*)d_in[17];
    const float* ln2b = (const float*)d_in[18];
    const float* w1   = (const float*)d_in[19];
    const float* b1   = (const float*)d_in[20];
    const float* w2   = (const float*)d_in[21];
    const float* b2   = (const float*)d_in[22];
    const float* lnfg = (const float*)d_in[23];
    const float* lnfb = (const float*)d_in[24];
    const float* logit_scale = (const float*)d_in[25];

    // Workspace carve-up (float units; every bf16 region starts 16B-aligned).
    float* ws   = (float*)d_ws;
    float* x    = ws; ws += (size_t)BT * Dd;        // residual stream (f32)
    float* xn   = ws; ws += (size_t)BT * Dd;        // LN output (f32, for gates)
    float* kbuf = ws; ws += (size_t)BT * Mm;
    float* qbuf = ws; ws += (size_t)BT * Mm;
    float* vbuf = ws; ws += (size_t)BT * Mm;
    float* gwb  = ws; ws += BT;
    float* gfb  = ws; ws += BT;
    __bf16* xn_bf  = (__bf16*)ws; ws += (size_t)BT * Dd  / 2;  // LN output (bf16 A operand)
    __bf16* rd_bf  = (__bf16*)ws; ws += (size_t)BT * Mm  / 2;  // scan readouts (bf16)
    __bf16* hid_bf = (__bf16*)ws; ws += (size_t)BT * HID / 2;  // MLP hidden (bf16 only)
    __bf16* wbf    = (__bf16*)ws; ws += (size_t)Vv * Dd  / 2;  // bf16 weight scratch (max = center)

    float* logits = (float*)d_out;
    float* states = logits + (size_t)BT * Vv;       // [L,B,M,M] follows logits

    embed_kernel<<<(BT * Dd + 255) / 256, 256, 0, stream>>>(idx, center, x, BT * Dd, Dd);

    auto cvt = [&](const float* src, __bf16* dst, size_t n) {
        const int n4 = (int)(n / 4);
        cvt_bf16_kernel<<<(n4 + 255) / 256, 256, 0, stream>>>(src, dst, n4);
    };
    auto gemm = [&](const __bf16* A, const __bf16* Wb, const float* bias, const float* res,
                    const float* sp, float sm, float* Cf, __bf16* Cbf,
                    int rows, int cols, int K, int act) {
        const int tiles = (rows / 32) * (cols / 64);
        gemm_bf16_nt<<<(tiles + 7) / 8, 256, 0, stream>>>(A, Wb, bias, res, sp, sm,
                                                          Cf, Cbf, rows, cols, K, act);
    };

    for (int l = 0; l < Ll; ++l) {
        const float* wk_l = wk + (size_t)l * Mm * Dd;
        const float* wq_l = wq + (size_t)l * Mm * Dd;
        const float* wv_l = wv + (size_t)l * Mm * Dd;
        const float* wo_l = wo + (size_t)l * Dd * Mm;
        const float* w1_l = w1 + (size_t)l * HID * Dd;
        const float* w2_l = w2 + (size_t)l * Dd * HID;

        layernorm_kernel<<<BT, 256, 0, stream>>>(x, ln1g + (size_t)l * Dd,
                                                 ln1b + (size_t)l * Dd, xn, xn_bf, Dd);

        cvt(wk_l, wbf, (size_t)Mm * Dd);
        gemm(xn_bf, wbf, bk + (size_t)l * Mm, nullptr, nullptr, 1.0f, kbuf, nullptr, BT, Mm, Dd, 0);
        cvt(wq_l, wbf, (size_t)Mm * Dd);
        gemm(xn_bf, wbf, bq + (size_t)l * Mm, nullptr, nullptr, 1.0f, qbuf, nullptr, BT, Mm, Dd, 0);
        cvt(wv_l, wbf, (size_t)Mm * Dd);
        gemm(xn_bf, wbf, bv + (size_t)l * Mm, nullptr, nullptr, 1.0f, vbuf, nullptr, BT, Mm, Dd, 2);

        l2norm_kernel<<<BT, 128, 0, stream>>>(kbuf);
        l2norm_kernel<<<BT, 128, 0, stream>>>(qbuf);
        gate_kernel<<<BT, 128, 0, stream>>>(xn, wgw + (size_t)l * Dd, bgw + l,
                                            wgf + (size_t)l * Dd, bgf + l, gwb, gfb, Dd);

        scan_kernel<<<Bb, 256, 0, stream>>>(kbuf, qbuf, vbuf, gwb, gfb, rd_bf,
                                            states + (size_t)l * Bb * Mm * Mm, Tt);

        // x += reads @ wo^T + bo   (residual fused, in place)
        cvt(wo_l, wbf, (size_t)Dd * Mm);
        gemm(rd_bf, wbf, bo + (size_t)l * Dd, x, nullptr, 1.0f, x, nullptr, BT, Dd, Mm, 0);

        layernorm_kernel<<<BT, 256, 0, stream>>>(x, ln2g + (size_t)l * Dd,
                                                 ln2b + (size_t)l * Dd, nullptr, xn_bf, Dd);
        cvt(w1_l, wbf, (size_t)HID * Dd);
        gemm(xn_bf, wbf, b1 + (size_t)l * HID, nullptr, nullptr, 1.0f,
             nullptr, hid_bf, BT, HID, Dd, 1);                    // gelu -> bf16 only
        cvt(w2_l, wbf, (size_t)Dd * HID);
        gemm(hid_bf, wbf, b2 + (size_t)l * Dd, x, nullptr, 1.0f, x, nullptr, BT, Dd, HID, 0);
    }

    layernorm_kernel<<<BT, 256, 0, stream>>>(x, lnfg, lnfb, nullptr, xn_bf, Dd);
    cvt(center, wbf, (size_t)Vv * Dd);
    // logits = (xf @ center^T) * logit_scale / sqrt(512)
    gemm(xn_bf, wbf, nullptr, nullptr, logit_scale, 0.04419417382415922f,
         logits, nullptr, BT, Vv, Dd, 0);
}
